// GraphSage_46050639348025
// MI455X (gfx1250) — matrix operand order repaired and checked by reading, not compiled, not run on hardware
//
#include <hip/hip_runtime.h>
#include <hip/hip_bf16.h>
#include <stdint.h>

// ---------------------------------------------------------------------------
// GraphSAGE inference (dead h1 eliminated):
//   agg2 = seg_mean(feat, neigh_idx[1], neigh_mask[1])         [B,256]
//   x2   = concat(agg2, hidden1[node_batch])                   [B,512]  -> bf16
//   h2   = LN(relu(x2 @ W2 + b2))                              [B,256]  -> bf16
//   out  = softmax(h2 @ Wc + bc)                               [B,64]   f32
// Sizes: N=200000 F=256 B=8192 S=10 H=256 C=64
// Memory-bound on the neighbor gather (~100 MB total ≈ 4.3 µs @ 23.3 TB/s);
// GEMMs (~2.4 GFLOP) run on the WMMA pipe with bf16 operands, weights L2-hot.
// W2 staging uses double-buffered GLOBAL_LOAD_ASYNC_TO_LDS_B128 so the DMA for
// K-step kk+1 overlaps the WMMAs of step kk (counted s_wait_asynccnt 0x4).
// ---------------------------------------------------------------------------

#define NB      8192
#define SS      10
#define FF      256
#define HH      256
#define DIN     512
#define CC      64
#define LN_EPS  1e-5f

typedef __attribute__((ext_vector_type(16))) __bf16 v16bf;
typedef __attribute__((ext_vector_type(8)))  float  v8f;

union Frag16 { v16bf v; unsigned int u[8]; };

__device__ __forceinline__ unsigned short f2bf(float f) {
    unsigned int u = __float_as_uint(f);
    unsigned int r = (u + 0x7FFFu + ((u >> 16) & 1u)) >> 16;  // RNE
    return (unsigned short)r;
}
__device__ __forceinline__ unsigned int pack2bf(float lo, float hi) {
    return (unsigned int)f2bf(lo) | ((unsigned int)f2bf(hi) << 16);
}

// Truncate a generic (flat) pointer into the 32-bit LDS offset expected by
// LDS-destination VGPR operands (shared aperture keeps the AS3 offset in the
// low 32 bits).
__device__ __forceinline__ unsigned lds_addr32(const void* p) {
    return (unsigned)(uintptr_t)p;
}

// ---------------------------------------------------------------------------
// Kernel 1: f32 -> bf16 weight conversion (W2, Wc). n must be even.
// ---------------------------------------------------------------------------
__global__ __launch_bounds__(256)
void k_f32_to_bf16(const float* __restrict__ src, unsigned int* __restrict__ dst, int n2) {
    int i = blockIdx.x * blockDim.x + threadIdx.x;
    if (i < n2) dst[i] = pack2bf(src[2 * i], src[2 * i + 1]);
}

// ---------------------------------------------------------------------------
// Kernel 2: build X = [agg2 | hidden1[node_batch]] in bf16, row-major [B,512].
// One wave per batch row; lane owns 8 contiguous columns (2x float4 loads per
// neighbor row -> fully coalesced 128B transactions per wave). The next
// neighbor row is prefetched (global_prefetch_b8) while the current one is
// accumulated to hide the index-dependent latency chain.
// ---------------------------------------------------------------------------
__global__ __launch_bounds__(256)
void k_build_x(const int* __restrict__ node_batch,
               const int* __restrict__ neigh_idx,            // full [2,B,S]
               const unsigned char* __restrict__ neigh_mask, // bool (1B) [2,B,S]
               const float* __restrict__ feat,               // [N,256]
               const float* __restrict__ hidden1,            // [N,256]
               unsigned int* __restrict__ X)                 // bf16 [B,512] as uints
{
    const int wave = threadIdx.x >> 5;
    const int lane = threadIdx.x & 31;
    const int row  = blockIdx.x * 8 + wave;
    if (row >= NB) return;

    const int* idx = neigh_idx + (1 * NB * SS) + row * SS;            // layer-2 hop
    const unsigned char* msk = neigh_mask + (1 * NB * SS) + row * SS;

    const int c0 = lane * 8;                       // my 8 columns
    float acc[8];
#pragma unroll
    for (int i = 0; i < 8; ++i) acc[i] = 0.0f;
    int cnt = 0;

    // prefetch first neighbor row
    __builtin_prefetch(feat + (size_t)idx[0] * FF + c0, 0, 1);

#pragma unroll
    for (int s = 0; s < SS; ++s) {
        if (s + 1 < SS)
            __builtin_prefetch(feat + (size_t)idx[s + 1] * FF + c0, 0, 1);
        if (msk[s]) {
            ++cnt;
            const float4* r4 = (const float4*)(feat + (size_t)idx[s] * FF + c0);
            float4 a = r4[0], b = r4[1];
            acc[0] += a.x; acc[1] += a.y; acc[2] += a.z; acc[3] += a.w;
            acc[4] += b.x; acc[5] += b.y; acc[6] += b.z; acc[7] += b.w;
        }
    }
    const float inv = 1.0f / (float)(cnt > 0 ? cnt : 1);

    // agg2 -> cols [0,256)
    uint4 o;
    o.x = pack2bf(acc[0] * inv, acc[1] * inv);
    o.y = pack2bf(acc[2] * inv, acc[3] * inv);
    o.z = pack2bf(acc[4] * inv, acc[5] * inv);
    o.w = pack2bf(acc[6] * inv, acc[7] * inv);
    ((uint4*)(X + (size_t)row * 256))[lane] = o;

    // hidden1[node_batch[row]] -> cols [256,512)
    const int self = node_batch[row];
    const float4* h4 = (const float4*)(hidden1 + (size_t)self * HH + c0);
    float4 a = h4[0], b = h4[1];
    uint4 p;
    p.x = pack2bf(a.x, a.y);
    p.y = pack2bf(a.z, a.w);
    p.z = pack2bf(b.x, b.y);
    p.w = pack2bf(b.z, b.w);
    ((uint4*)(X + (size_t)row * 256 + 128))[lane] = p;
}

// ---------------------------------------------------------------------------
// WMMA fragment loaders.
//
// A (16x32 bf16, M x K), ISA layout: lane<16 -> M=lane, K in {kb..kb+7} U
// {kb+16..kb+23} with kb=0; lane>=16 -> same M, kb=8. Packed K pairs sit at
// even K -> two contiguous uint4 (b128) loads per fragment.
//
// B (32x16 bf16, K x N): lane -> N = lane%16, elements follow the same K
// order as A (table order) -> strided 16-bit gathers down the K rows.
// ---------------------------------------------------------------------------
__device__ __forceinline__ void load_a_frag(Frag16& f, const unsigned int* rowbase /* bf16 row as uints */,
                                            int k0_uints /* K0/2 */, int lane) {
    const int kb = k0_uints + ((lane >> 4) << 2);   // +0 or +4 uints (K +0/+8)
    const uint4* p = (const uint4*)(rowbase + kb);
    uint4 q0 = p[0];
    uint4 q1 = p[2];                                 // +8 uints (K +16)
    f.u[0] = q0.x; f.u[1] = q0.y; f.u[2] = q0.z; f.u[3] = q0.w;
    f.u[4] = q1.x; f.u[5] = q1.y; f.u[6] = q1.z; f.u[7] = q1.w;
}

__device__ __forceinline__ void load_b_frag(Frag16& f, const unsigned int* base,
                                            int row_stride_uints, int nb_uints, int lane) {
    const unsigned short* b16 = (const unsigned short*)base;
    const int n = 2 * nb_uints + (lane & 15);
    const int kh = (lane >> 4) << 3;
    const int st = 2 * row_stride_uints;
#pragma unroll
    for (int v = 0; v < 8; ++v) {
        const int k = kh + (v & 3) * 2 + ((v >> 2) << 4);
        f.u[v] = (unsigned int)b16[k * st + n] | ((unsigned int)b16[(k + 1) * st + n] << 16);
    }
}

// Issue the 4 per-thread async DMA b128 loads staging one 32x256 bf16 W tile.
__device__ __forceinline__ void stage_w_async(const unsigned int* wk /* uniform */,
                                              uint4* buf, int tid) {
#pragma unroll
    for (int j = 0; j < 4; ++j) {
        const unsigned elem = (unsigned)(tid + j * 256);     // uint4 index
        const unsigned goff = elem * 16u;                    // byte offset
        const unsigned ldsa = lds_addr32(buf + elem);
        asm volatile("global_load_async_to_lds_b128 %0, %1, %2"
                     :: "v"(ldsa), "v"(goff), "s"(wk)
                     : "memory");
    }
}

// ---------------------------------------------------------------------------
// Kernel 3: H2 = LN(relu(X @ W2 + b2)), X:[B,512]bf16 W2:[512,256]bf16.
// Block = 256 threads (8 waves) owns 16 rows x all 256 cols (wave w -> cols
// [32w,32w+32), two 16x16 WMMA tiles). W2 K-slices (16 KB each) are staged
// into double-buffered LDS with GLOBAL_LOAD_ASYNC_TO_LDS_B128; the counted
// wait (s_wait_asynccnt 4) keeps the next tile's DMA in flight behind the
// current tile's WMMAs. Epilogue LayerNorm via LDS + wave32 shuffles.
// ---------------------------------------------------------------------------
__global__ __launch_bounds__(256)
void k_gemm_ln(const unsigned int* __restrict__ X,   // bf16 [B,512]
               const unsigned int* __restrict__ W,   // bf16 [512,256]
               const float* __restrict__ bias,
               const float* __restrict__ gamma,
               const float* __restrict__ beta,
               unsigned short* __restrict__ H2)      // bf16 [B,256]
{
    __shared__ unsigned int sW[2][32 * 128];         // 2 x (32 K-rows x 256 bf16) = 32 KB
    __shared__ float        sO[16 * 256];            // f32 out tile              = 16 KB

    const int tid  = threadIdx.x;
    const int lane = tid & 31;
    const int wave = tid >> 5;
    const int m0   = blockIdx.x * 16;
    const int myrow = m0 + (lane & 15);
    const unsigned int* myrowptr = X + (size_t)myrow * 256;   // 512 bf16 = 256 uints

    v8f c0 = {0.f, 0.f, 0.f, 0.f, 0.f, 0.f, 0.f, 0.f};
    v8f c1 = {0.f, 0.f, 0.f, 0.f, 0.f, 0.f, 0.f, 0.f};

    // prologue: stage K-tile 0 into buffer 0
    stage_w_async(W, (uint4*)sW[0], tid);

    for (int kk = 0; kk < 16; ++kk) {                // K = 512, step 32
        const int K0 = kk * 32;
        const unsigned int* sWc = sW[kk & 1];

        if (kk + 1 < 16) {
            // issue DMA for the next tile into the other buffer, then wait
            // only for the OLDER group of 4 (in-order completion).
            stage_w_async(W + (kk + 1) * 32 * 128, (uint4*)sW[(kk + 1) & 1], tid);
            asm volatile("s_wait_asynccnt 0x4" ::: "memory");
        } else {
            asm volatile("s_wait_asynccnt 0x0" ::: "memory");
        }
        __syncthreads();

        Frag16 a;
        load_a_frag(a, myrowptr, K0 / 2, lane);

        Frag16 b0, b1;
        load_b_frag(b0, sWc, 128, (wave * 32) / 2, lane);        // cols 32w..+15
        load_b_frag(b1, sWc, 128, (wave * 32 + 16) / 2, lane);   // cols 32w+16..+31

        c0 = __builtin_amdgcn_wmma_f32_16x16x32_bf16(false, a.v, false, b0.v,
                                                     (short)0, c0, false, false);
        c1 = __builtin_amdgcn_wmma_f32_16x16x32_bf16(false, a.v, false, b1.v,
                                                     (short)0, c1, false, false);
        __syncthreads();
    }

    // epilogue: bias + relu -> LDS tile [16][256]
    {
        const int r_add = (lane >> 4) << 3;          // +0 or +8
        const int ncol  = lane & 15;
#pragma unroll
        for (int r = 0; r < 8; ++r) {
            int row = r + r_add;
            int col0 = wave * 32 + ncol;
            int col1 = wave * 32 + 16 + ncol;
            float v0 = c0[r] + bias[col0];
            float v1 = c1[r] + bias[col1];
            sO[row * 256 + col0] = v0 > 0.f ? v0 : 0.f;
            sO[row * 256 + col1] = v1 > 0.f ? v1 : 0.f;
        }
    }
    __syncthreads();

    // LayerNorm: wave w handles rows 2w, 2w+1 (wave32 shuffle reduction)
#pragma unroll
    for (int rr = wave * 2; rr < wave * 2 + 2; ++rr) {
        float sum = 0.f, sq = 0.f;
#pragma unroll
        for (int i = 0; i < 8; ++i) {
            float v = sO[rr * 256 + lane + 32 * i];
            sum += v; sq += v * v;
        }
        for (int off = 16; off > 0; off >>= 1) {
            sum += __shfl_down(sum, off);
            sq  += __shfl_down(sq,  off);
        }
        sum = __shfl(sum, 0);
        sq  = __shfl(sq, 0);
        const float mean = sum * (1.0f / 256.0f);
        const float var  = sq * (1.0f / 256.0f) - mean * mean;
        const float rstd = rsqrtf(var + LN_EPS);
#pragma unroll
        for (int i = 0; i < 8; ++i) {
            int col = lane + 32 * i;
            float v = sO[rr * 256 + col];
            float o = gamma[col] * (v - mean) * rstd + beta[col];
            H2[(size_t)(m0 + rr) * 256 + col] = f2bf(o);
        }
    }
}

// ---------------------------------------------------------------------------
// Kernel 4: out = softmax(H2 @ Wc + bc). Block = 128 threads (4 waves) owns
// 16 rows x 64 cols; wave w -> 16x16 tile at cols 16w. K = 256, step 32.
// ---------------------------------------------------------------------------
__global__ __launch_bounds__(128)
void k_gemm_softmax(const unsigned int* __restrict__ H2, // bf16 [B,256]
                    const unsigned int* __restrict__ Wc, // bf16 [256,64]
                    const float* __restrict__ bc,
                    float* __restrict__ out)             // [B,64]
{
    __shared__ float sS[16 * 64];

    const int lane = threadIdx.x & 31;
    const int wave = threadIdx.x >> 5;
    const int m0   = blockIdx.x * 16;
    const int myrow = m0 + (lane & 15);
    const unsigned int* myrowptr = H2 + (size_t)myrow * 128;  // 256 bf16 = 128 uints
    const int n0 = wave * 16;

    v8f c = {0.f, 0.f, 0.f, 0.f, 0.f, 0.f, 0.f, 0.f};

#pragma unroll
    for (int kk = 0; kk < 8; ++kk) {                 // K = 256, step 32
        const int K0 = kk * 32;
        Frag16 a;
        load_a_frag(a, myrowptr, K0 / 2, lane);
        Frag16 b;
        load_b_frag(b, Wc + K0 * 32, 32, n0 / 2, lane);  // Wc row = 64 bf16 = 32 uints
        c = __builtin_amdgcn_wmma_f32_16x16x32_bf16(false, a.v, false, b.v,
                                                    (short)0, c, false, false);
    }

    {
        const int r_add = (lane >> 4) << 3;
        const int col   = n0 + (lane & 15);
#pragma unroll
        for (int r = 0; r < 8; ++r)
            sS[(r + r_add) * 64 + col] = c[r] + bc[col];
    }
    __syncthreads();

    // softmax: wave w handles rows 4w..4w+3; 64 cols -> 2 values/lane
#pragma unroll
    for (int rr = wave * 4; rr < wave * 4 + 4; ++rr) {
        float v0 = sS[rr * 64 + lane];
        float v1 = sS[rr * 64 + lane + 32];
        float mx = fmaxf(v0, v1);
        for (int off = 16; off > 0; off >>= 1)
            mx = fmaxf(mx, __shfl_down(mx, off));
        mx = __shfl(mx, 0);
        float e0 = __expf(v0 - mx);
        float e1 = __expf(v1 - mx);
        float s = e0 + e1;
        for (int off = 16; off > 0; off >>= 1)
            s += __shfl_down(s, off);
        s = __shfl(s, 0);
        const float inv = 1.0f / s;
        out[(size_t)(m0 + rr) * 64 + lane]      = e0 * inv;
        out[(size_t)(m0 + rr) * 64 + lane + 32] = e1 * inv;
    }
}

// ---------------------------------------------------------------------------
extern "C" void kernel_launch(void* const* d_in, const int* in_sizes, int n_in,
                              void* d_out, int out_size, void* d_ws, size_t ws_size,
                              hipStream_t stream) {
    const int*           node_batch = (const int*)d_in[0];
    const int*           neigh_idx  = (const int*)d_in[1];
    const unsigned char* neigh_mask = (const unsigned char*)d_in[2]; // bool array
    const float*         feat       = (const float*)d_in[3];
    const float*         hidden1    = (const float*)d_in[4];
    // d_in[5..8] = W1,b1,g1,be1 : dead code in the reference (h1 unused)
    const float*         W2 = (const float*)d_in[9];
    const float*         b2 = (const float*)d_in[10];
    const float*         g2 = (const float*)d_in[11];
    const float*         be2 = (const float*)d_in[12];
    const float*         Wc = (const float*)d_in[13];
    const float*         bc = (const float*)d_in[14];
    float*               out = (float*)d_out;

    // workspace layout (bf16 payloads stored as ushort/uint)
    unsigned short* wsp  = (unsigned short*)d_ws;
    unsigned short* X    = wsp;                      // [B,512]   4194304 elems
    unsigned short* W2b  = X + (size_t)NB * DIN;     // [512,256]  131072
    unsigned short* Wcb  = W2b + (size_t)DIN * HH;   // [256,64]    16384
    unsigned short* H2   = Wcb + (size_t)HH * CC;    // [B,256]   2097152

    // 1) weight conversion
    k_f32_to_bf16<<<(DIN * HH / 2 + 255) / 256, 256, 0, stream>>>(W2, (unsigned int*)W2b, DIN * HH / 2);
    k_f32_to_bf16<<<(HH * CC / 2 + 255) / 256, 256, 0, stream>>>(Wc, (unsigned int*)Wcb, HH * CC / 2);

    // 2) gather/aggregate + concat -> X (bf16)
    k_build_x<<<NB / 8, 256, 0, stream>>>(node_batch, neigh_idx, neigh_mask,
                                          feat, hidden1, (unsigned int*)X);

    // 3) fused GEMM + bias + relu + LayerNorm -> H2 (bf16)
    k_gemm_ln<<<NB / 16, 256, 0, stream>>>((const unsigned int*)X, (const unsigned int*)W2b,
                                           b2, g2, be2, H2);

    // 4) fused classifier GEMM + softmax -> out (f32)
    k_gemm_softmax<<<NB / 16, 128, 0, stream>>>((const unsigned int*)H2, (const unsigned int*)Wcb,
                                                bc, out);
    (void)in_sizes; (void)n_in; (void)out_size; (void)ws_size;
}
